// FineMatchingModule_51367808860500
// MI455X (gfx1250) — compile-verified, matrix operand order checked by reading, not run on hardware
//
#include <hip/hip_runtime.h>
#include <hip/hip_bf16.h>

// MI455X / gfx1250, wave32, WMMA 16x16x32 bf16 with f32 accumulate.

typedef __attribute__((ext_vector_type(16))) __bf16 v16bf;
typedef __attribute__((ext_vector_type(8)))  float  v8f;

#define BB 4
#define KK 1024
#define CC 256
#define HH 256
#define WW 256
#define FWIN 5
#define N25 25
#define ITEMS (BB*KK)      // 4096
#define MATEL (CC*CC)      // 65536 elements per packed weight matrix

static __device__ __forceinline__ unsigned short f2bf(float f) {
    unsigned u = __builtin_bit_cast(unsigned, f);
    unsigned r = u + 0x7fffu + ((u >> 16) & 1u);   // round-to-nearest-even
    return (unsigned short)(r >> 16);
}

// ---------------------------------------------------------------------------
// Pack weights into WMMA B-fragment order (bf16). Matrix m: l=m/4,
// t=m%4 -> {Wq, Wk^T, Wv, Wo}. Element order matches the GEMM's contiguous
// per-lane uint4 loads: idx = ((tn*8+kc)*32 + lane)*16 + j.
// ---------------------------------------------------------------------------
__global__ void pack_weights(const float* __restrict__ Wq, const float* __restrict__ Wk,
                             const float* __restrict__ Wv, const float* __restrict__ Wo,
                             unsigned short* __restrict__ Wp) {
    int tid = blockIdx.x * blockDim.x + threadIdx.x;
    if (tid >= 8 * MATEL) return;
    int m    = tid >> 16;
    int rix  = tid & 65535;
    int tn   = rix >> 12;
    int rem  = rix & 4095;
    int kc   = rem >> 9;
    int rem2 = rem & 511;
    int lane = rem2 >> 4;
    int j    = rem2 & 15;
    int g = j >> 1, e = j & 1, half = lane >> 4;
    int n = tn * 16 + (lane & 15);
    int k = kc * 32 + 2 * g + 8 * half + ((g >= 4) ? 8 : 0) + e;
    int l = m >> 2, t = m & 3;
    const float* W = (t == 0 ? Wq : t == 1 ? Wk : t == 2 ? Wv : Wo) + (size_t)l * MATEL;
    float val = (t == 1) ? W[n * CC + k] : W[k * CC + n];   // t==1 -> Wk^T
    Wp[tid] = f2bf(val);
}

__global__ void zero_f32(float* __restrict__ p, int n) {
    int i = blockIdx.x * blockDim.x + threadIdx.x;
    if (i < n) p[i] = 0.f;
}

// ---------------------------------------------------------------------------
// Gather: per (b,k) item, extract center vector S and the 5x5 window with the
// reference's channel-major reshape quirk: X[item][j] with j = ch*25 + sp,
// later read as X[item][n*256 + c]. Staged through LDS: coalesced 1KB-per-
// position loads, scatter into LDS (stride 25, conflict-free), coalesced out.
// ---------------------------------------------------------------------------
__global__ void gather_kernel(const float* __restrict__ tf, const float* __restrict__ pos,
                              float* __restrict__ Xf, float* __restrict__ Sf,
                              unsigned short* __restrict__ Sh) {
    __shared__ float win[N25 * CC];
    int item = blockIdx.x;
    int b = item / KK;
    int t = threadIdx.x;
    int r = (int)pos[item * 2 + 0];
    int c = (int)pos[item * 2 + 1];
    const float* fb = tf + (size_t)b * HH * WW * CC;

    float sv = fb[((size_t)r * WW + c) * CC + t];
    Sf[(size_t)item * CC + t] = sv;
    Sh[(size_t)item * CC + t] = f2bf(sv);

    for (int sp = 0; sp < N25; ++sp) {
        int dy = sp / FWIN, dx = sp % FWIN;
        int rr = r + dy - 2, cc2 = c + dx - 2;
        float v = 0.f;
        if (rr >= 0 && rr < HH && cc2 >= 0 && cc2 < WW)
            v = fb[((size_t)rr * WW + cc2) * CC + t];
        win[t * N25 + sp] = v;   // channel-major flat index: ch*25+sp
    }
    __syncthreads();
    float* xo = Xf + (size_t)item * (N25 * CC);
    for (int j = t; j < N25 * CC; j += blockDim.x) xo[j] = win[j];
}

// ---------------------------------------------------------------------------
// Generic GEMM: C(4096,256) = A(4096,256)bf16 @ Bpacked(256,256)bf16 [+ Cin].
// 8 waves/block, one 16x16 tile per wave, 8 WMMAs over K=256.
// ---------------------------------------------------------------------------
__global__ void gemm_bf16(const unsigned short* __restrict__ A,
                          const unsigned short* __restrict__ Bp,
                          float* __restrict__ outF,
                          unsigned short* __restrict__ outH,
                          const float* __restrict__ Cin) {
    int lane = threadIdx.x & 31;
    int wave = threadIdx.x >> 5;
    int wid  = blockIdx.x * 8 + wave;   // 0..4095 tiles
    int tm   = wid >> 4;                // 0..255  (M tiles)
    int tn   = wid & 15;                // 0..15   (N tiles)
    int half = lane >> 4;
    int mr   = lane & 15;
    const unsigned short* ap = A + (size_t)(tm * 16 + mr) * CC;

    v8f acc;
    #pragma unroll
    for (int v = 0; v < 8; ++v)
        acc[v] = Cin ? Cin[(size_t)(tm * 16 + v + 8 * half) * CC + tn * 16 + mr] : 0.f;

    #pragma unroll
    for (int kc = 0; kc < 8; ++kc) {
        union { v16bf v; unsigned int u[8]; } a;
        int kbase = kc * 32 + 8 * half;
        #pragma unroll
        for (int g = 0; g < 8; ++g) {
            int k = kbase + 2 * g + ((g >= 4) ? 8 : 0);
            a.u[g] = *(const unsigned int*)(ap + k);
        }
        union { v16bf v; uint4 q[2]; } b;
        const uint4* bp = (const uint4*)(Bp + ((size_t)(tn * 8 + kc) * 32 + lane) * 16);
        b.q[0] = bp[0];
        b.q[1] = bp[1];
        acc = __builtin_amdgcn_wmma_f32_16x16x32_bf16(false, a.v, false, b.v,
                                                      (short)0, acc, false, false);
    }

    int col = tn * 16 + mr;
    #pragma unroll
    for (int v = 0; v < 8; ++v) {
        size_t idx = (size_t)(tm * 16 + v + 8 * half) * CC + col;
        float f = acc[v];
        if (outF) outF[idx] = f;
        if (outH) outH[idx] = f2bf(f);
    }
}

// ---------------------------------------------------------------------------
// Per-item attention core: X_l[n] = X0[n] + Macc; scores[n] = X_l[n].qh / 16;
// softmax over 25; Y = sum_n attn[n] * X_l[n]  -> bf16 (feeds Wv/Wo GEMMs).
// ---------------------------------------------------------------------------
__global__ void attn_kernel(const float* __restrict__ Xf, const float* __restrict__ Macc,
                            const float* __restrict__ qhf, unsigned short* __restrict__ Yh) {
    __shared__ float xl[N25 * CC];
    __shared__ float red[N25 * CC];
    __shared__ float sc[32];
    int item = blockIdx.x;
    int t = threadIdx.x;
    float mv = Macc[(size_t)item * CC + t];
    float qv = qhf[(size_t)item * CC + t];
    const float* xi = Xf + (size_t)item * (N25 * CC);
    for (int n = 0; n < N25; ++n) {
        float x = xi[n * CC + t] + mv;
        xl[n * CC + t] = x;
        red[n * CC + t] = x * qv;
    }
    __syncthreads();
    for (int off = 128; off > 0; off >>= 1) {
        if (t < off)
            for (int n = 0; n < N25; ++n)
                red[n * CC + t] += red[n * CC + t + off];
        __syncthreads();
    }
    if (t < N25) sc[t] = red[t * CC] * 0.0625f;   // 1/sqrt(256)
    __syncthreads();
    float mx = -1e30f;
    for (int n = 0; n < N25; ++n) mx = fmaxf(mx, sc[n]);
    float s = 0.f, w[N25];
    for (int n = 0; n < N25; ++n) { w[n] = __expf(sc[n] - mx); s += w[n]; }
    float inv = 1.f / s;
    float y = 0.f;
    for (int n = 0; n < N25; ++n) y += w[n] * inv * xl[n * CC + t];
    Yh[(size_t)item * CC + t] = f2bf(y);
}

// ---------------------------------------------------------------------------
// Final: corr[n] = S.(X0[n]+Macc); softmax; refined = pos + prob @ offsets.
// offsets from linspace(-3,2,5) meshgrid('ij'), faithful to the reference.
// ---------------------------------------------------------------------------
__global__ void final_kernel(const float* __restrict__ Xf, const float* __restrict__ Macc,
                             const float* __restrict__ Sf, const float* __restrict__ pos,
                             float* __restrict__ out) {
    __shared__ float red[N25 * CC];
    __shared__ float sc[32];
    int item = blockIdx.x;
    int t = threadIdx.x;
    float mv = Macc[(size_t)item * CC + t];
    float sv = Sf[(size_t)item * CC + t];
    const float* xi = Xf + (size_t)item * (N25 * CC);
    for (int n = 0; n < N25; ++n)
        red[n * CC + t] = (xi[n * CC + t] + mv) * sv;
    __syncthreads();
    for (int off = 128; off > 0; off >>= 1) {
        if (t < off)
            for (int n = 0; n < N25; ++n)
                red[n * CC + t] += red[n * CC + t + off];
        __syncthreads();
    }
    if (t < N25) sc[t] = red[t * CC];
    __syncthreads();
    float mx = -1e30f;
    for (int n = 0; n < N25; ++n) mx = fmaxf(mx, sc[n]);
    float s = 0.f, w[N25];
    for (int n = 0; n < N25; ++n) { w[n] = __expf(sc[n] - mx); s += w[n]; }
    float inv = 1.f / s;
    const float off1d[5] = {-3.f, -1.75f, -0.5f, 0.75f, 2.f};
    float dr = 0.f, dc = 0.f;
    for (int n = 0; n < N25; ++n) {
        float p = w[n] * inv;
        dr += p * off1d[n / 5];
        dc += p * off1d[n % 5];
    }
    if (t == 0) {
        out[item * 2 + 0] = pos[item * 2 + 0] + dr;
        out[item * 2 + 1] = pos[item * 2 + 1] + dc;
    }
}

extern "C" void kernel_launch(void* const* d_in, const int* in_sizes, int n_in,
                              void* d_out, int out_size, void* d_ws, size_t ws_size,
                              hipStream_t stream) {
    (void)in_sizes; (void)n_in; (void)out_size; (void)ws_size;
    const float* tf  = (const float*)d_in[1];   // target_features (source_features unused)
    const float* pos = (const float*)d_in[2];
    const float* Wq  = (const float*)d_in[3];
    const float* Wk  = (const float*)d_in[4];
    const float* Wv  = (const float*)d_in[5];
    const float* Wo  = (const float*)d_in[6];
    float* out = (float*)d_out;

    char* w = (char*)d_ws;
    size_t o = 0;
    float* Xf   = (float*)(w + o); o += (size_t)ITEMS * N25 * CC * 4;   // 104.9 MB
    float* Sf   = (float*)(w + o); o += (size_t)ITEMS * CC * 4;
    float* Macc = (float*)(w + o); o += (size_t)ITEMS * CC * 4;
    float* qhf  = (float*)(w + o); o += (size_t)ITEMS * CC * 4;
    unsigned short* Sh = (unsigned short*)(w + o); o += (size_t)ITEMS * CC * 2;
    unsigned short* Qh = (unsigned short*)(w + o); o += (size_t)ITEMS * CC * 2;
    unsigned short* Yh = (unsigned short*)(w + o); o += (size_t)ITEMS * CC * 2;
    unsigned short* Th = (unsigned short*)(w + o); o += (size_t)ITEMS * CC * 2;
    unsigned short* Wp = (unsigned short*)(w + o); o += (size_t)8 * MATEL * 2;

    pack_weights<<<2048, 256, 0, stream>>>(Wq, Wk, Wv, Wo, Wp);
    zero_f32<<<(ITEMS * CC) / 256, 256, 0, stream>>>(Macc, ITEMS * CC);
    gather_kernel<<<ITEMS, 256, 0, stream>>>(tf, pos, Xf, Sf, Sh);

    for (int l = 0; l < 2; ++l) {
        const unsigned short* wl = Wp + (size_t)l * 4 * MATEL;
        // qh = (S @ Wq) @ Wk^T
        gemm_bf16<<<512, 256, 0, stream>>>(Sh, wl + 0 * MATEL, nullptr, Qh, nullptr);
        gemm_bf16<<<512, 256, 0, stream>>>(Qh, wl + 1 * MATEL, qhf, nullptr, nullptr);
        // per-item scores + softmax + weighted sum
        attn_kernel<<<ITEMS, 256, 0, stream>>>(Xf, Macc, qhf, Yh);
        // Macc += (Y @ Wv) @ Wo
        gemm_bf16<<<512, 256, 0, stream>>>(Yh, wl + 2 * MATEL, nullptr, Th, nullptr);
        gemm_bf16<<<512, 256, 0, stream>>>(Th, wl + 3 * MATEL, Macc, nullptr, Macc);
    }

    final_kernel<<<ITEMS, 256, 0, stream>>>(Xf, Macc, Sf, pos, out);
}
